// MHA_42314017800203
// MI455X (gfx1250) — matrix operand order repaired and checked by reading, not compiled, not run on hardware
//
#include <hip/hip_runtime.h>

// ---------------- types ----------------
typedef __attribute__((ext_vector_type(16))) __bf16 v16bf;
typedef __attribute__((ext_vector_type(8)))  float  v8f;
typedef __attribute__((ext_vector_type(4)))  unsigned int u32x4;
typedef __attribute__((ext_vector_type(4)))  int v4i;

#define BB   2
#define SS   2048
#define DD   2048
#define HH   16
#define DKK  128
#define E3D  (3*DD)          // 6144
#define MROWS (BB*SS)        // 4096

#if defined(__has_builtin)
#if __has_builtin(__builtin_amdgcn_global_load_async_to_lds_b128)
#define USE_ASYNC_LDS 1
#endif
#endif

__device__ __forceinline__ unsigned short f2bf(float f) {
  union { float f; unsigned u; } v; v.f = f;
  unsigned r = v.u + 0x7FFFu + ((v.u >> 16) & 1u);   // RNE
  return (unsigned short)(r >> 16);
}

// A operand: 16x32 bf16, lane r=lane&15 -> row r; kb=(lane>>4)*8;
// halves 0..7 = K kb..kb+7, halves 8..15 = K kb+16..kb+23   (ISA 7.12.2)
__device__ __forceinline__ v16bf packA(const unsigned short* base, int ld, int lane) {
  int r = lane & 15, kb = (lane >> 4) << 3;
  const unsigned short* p = base + r * ld + kb;
  union { v16bf v; u32x4 q[2]; } u;
  u.q[0] = *(const u32x4*)(p);
  u.q[1] = *(const u32x4*)(p + 16);
  return u.v;
}

// B operand: 32x16 bf16 fed from B^T rows (row n = output column n, K contiguous);
// lane n=lane&15 -> column n; kb=(lane>>4)*8; halves 0..7 = K kb..kb+7, halves 8..15 = K kb+16..kb+23
__device__ __forceinline__ v16bf packB(const unsigned short* baseT, long long ld, int lane) {
  int n = lane & 15, ko = (lane >> 4) << 3;
  const unsigned short* p = baseT + (long long)n * ld + ko;
  union { v16bf v; u32x4 q[2]; } u;
  u.q[0] = *(const u32x4*)(p);
  u.q[1] = *(const u32x4*)(p + 16);
  return u.v;
}

__device__ __forceinline__ v8f wmma_bf16(v16bf a, v16bf b, v8f c) {
  return __builtin_amdgcn_wmma_f32_16x16x32_bf16(false, a, false, b, (short)0, c, false, false);
}

// stage 16B global -> LDS (async copy path when available)
__device__ __forceinline__ void stage16(const unsigned short* g, unsigned short* l) {
#ifdef USE_ASYNC_LDS
  __builtin_amdgcn_global_load_async_to_lds_b128(
      (__attribute__((address_space(1))) v4i*)g,
      (__attribute__((address_space(3))) v4i*)l, 0, 0);
#else
  *(u32x4*)l = *(const u32x4*)g;
#endif
}

__device__ __forceinline__ void stage_fence() {
#ifdef USE_ASYNC_LDS
  asm volatile("s_wait_asynccnt 0x0" ::: "memory");
#endif
}

// ---------------- f32 -> bf16 convert ----------------
__global__ void cvt_bf16_kernel(const float* __restrict__ src,
                                unsigned short* __restrict__ dst, int n) {
  int i = blockIdx.x * blockDim.x + threadIdx.x;
  int stride = gridDim.x * blockDim.x;
  for (; i < n; i += stride) dst[i] = f2bf(src[i]);
}

// ---------------- shared GEMM core ----------------
// Block tile: 256(M) x 128(N), 8 waves in 4x2 grid, each wave 64x64 (16 WMMA tiles).
#define LDT 40   // padded LDS row stride (bf16 elems)

// computes acc[4][4] for this wave; A rows = M, B rows = N (weight rows, K contiguous)
__device__ __forceinline__ void gemm_core(
    const unsigned short* __restrict__ Amat, const unsigned short* __restrict__ Bmat,
    int Mbase, int Nbase, int tid, int lane, int wr, int wc,
    unsigned short* At, unsigned short* Bt, v8f acc[4][4]) {
#pragma unroll
  for (int mt = 0; mt < 4; ++mt)
#pragma unroll
    for (int nt = 0; nt < 4; ++nt) acc[mt][nt] = (v8f)0.0f;

  for (int k0 = 0; k0 < DD; k0 += 32) {
    __syncthreads();                       // everyone done with previous tile
#pragma unroll
    for (int j = 0; j < 4; ++j) {          // A: 256 rows x 4 col-groups = 1024 segs
      int seg = tid + 256 * j;
      int row = seg >> 2, cg = (seg & 3) << 3;
      stage16(&Amat[(size_t)(Mbase + row) * DD + k0 + cg], &At[row * LDT + cg]);
    }
#pragma unroll
    for (int j = 0; j < 2; ++j) {          // B: 128 rows x 4 col-groups = 512 segs
      int seg = tid + 256 * j;
      int row = seg >> 2, cg = (seg & 3) << 3;
      stage16(&Bmat[(size_t)(Nbase + row) * DD + k0 + cg], &Bt[row * LDT + cg]);
    }
    stage_fence();
    __syncthreads();                       // all LDS writes visible
    v16bf a[4], b[4];
#pragma unroll
    for (int mt = 0; mt < 4; ++mt) a[mt] = packA(&At[(wr * 64 + mt * 16) * LDT], LDT, lane);
#pragma unroll
    for (int nt = 0; nt < 4; ++nt) b[nt] = packB(&Bt[(wc * 64 + nt * 16) * LDT], LDT, lane);
#pragma unroll
    for (int mt = 0; mt < 4; ++mt)
#pragma unroll
      for (int nt = 0; nt < 4; ++nt) acc[mt][nt] = wmma_bf16(a[mt], b[nt], acc[mt][nt]);
  }
}

// ---------------- QKV projection GEMM ----------------
__global__ __launch_bounds__(256) void qkv_gemm_kernel(
    const unsigned short* __restrict__ xb, const unsigned short* __restrict__ wq,
    const float* __restrict__ bqkv,
    unsigned short* __restrict__ Qb, unsigned short* __restrict__ Kb,
    unsigned short* __restrict__ vTb) {
  __shared__ unsigned short At[256 * LDT];
  __shared__ unsigned short Bt[128 * LDT];
  const int tid = threadIdx.x, lane = tid & 31, w = tid >> 5;
  const int wr = w >> 1, wc = w & 1;                    // 4x2 wave grid
  const int Mbase = blockIdx.y * 256, Nbase = blockIdx.x * 128;

  v8f acc[4][4];
  gemm_core(xb, wq, Mbase, Nbase, tid, lane, wr, wc, At, Bt, acc);

  const int halfsel = lane >> 4, nloc = lane & 15;
#pragma unroll
  for (int nt = 0; nt < 4; ++nt) {
    int n = Nbase + wc * 64 + nt * 16 + nloc;           // 0..6143
    float bias = bqkv[n];
    int h = n / 384, r = n - h * 384;
    int comp = r >> 7, dk = r & 127;
#pragma unroll
    for (int mt = 0; mt < 4; ++mt) {
#pragma unroll
      for (int i = 0; i < 8; ++i) {
        int m = Mbase + wr * 64 + mt * 16 + i + 8 * halfsel;
        int bb = m >> 11, s = m & 2047;
        int bh = bb * HH + h;
        float val = acc[mt][nt][i] + bias;
        if (comp == 0) {
          val *= 0.08838834764831845f;                  // 1/sqrt(DK) folded into Q
          Qb[((size_t)bh * SS + s) * DKK + dk] = f2bf(val);
        } else if (comp == 1) {
          Kb[((size_t)bh * SS + s) * DKK + dk] = f2bf(val);
        } else {
          vTb[((size_t)bh * DKK + dk) * SS + s] = f2bf(val);
        }
      }
    }
  }
}

// ---------------- output projection GEMM ----------------
__global__ __launch_bounds__(256) void oproj_gemm_kernel(
    const unsigned short* __restrict__ ob, const unsigned short* __restrict__ wo,
    const float* __restrict__ bo, float* __restrict__ out) {
  __shared__ unsigned short At[256 * LDT];
  __shared__ unsigned short Bt[128 * LDT];
  const int tid = threadIdx.x, lane = tid & 31, w = tid >> 5;
  const int wr = w >> 1, wc = w & 1;
  const int Mbase = blockIdx.y * 256, Nbase = blockIdx.x * 128;

  v8f acc[4][4];
  gemm_core(ob, wo, Mbase, Nbase, tid, lane, wr, wc, At, Bt, acc);

  const int halfsel = lane >> 4, nloc = lane & 15;
#pragma unroll
  for (int nt = 0; nt < 4; ++nt) {
    int n = Nbase + wc * 64 + nt * 16 + nloc;
    float bias = bo[n];
#pragma unroll
    for (int mt = 0; mt < 4; ++mt) {
#pragma unroll
      for (int i = 0; i < 8; ++i) {
        int m = Mbase + wr * 64 + mt * 16 + i + 8 * halfsel;
        out[(size_t)m * DD + n] = acc[mt][nt][i] + bias;
      }
    }
  }
}

// ---------------- flash attention ----------------
// grid: (SS/128, BB*HH); 8 waves * 16 queries each; 64-key tiles; online softmax
#define PLD 72   // 16x64 staging, padded
__global__ __launch_bounds__(256) void attn_kernel(
    const unsigned short* __restrict__ Qb, const unsigned short* __restrict__ Kb,
    const unsigned short* __restrict__ vTb, unsigned short* __restrict__ Ob) {
  __shared__ unsigned short Plds[8][16 * PLD];
  const int tid = threadIdx.x, lane = tid & 31, w = tid >> 5;
  const int halfsel = lane >> 4, nloc = lane & 15;
  const int bh = blockIdx.y, bb = bh >> 4, h = bh & 15;
  const int q0 = blockIdx.x * 128 + w * 16;

  const unsigned short* Qp  = Qb  + (size_t)bh * SS * DKK;
  const unsigned short* Kp  = Kb  + (size_t)bh * SS * DKK;
  const unsigned short* vTp = vTb + (size_t)bh * DKK * SS;

  // pre-pack this wave's 16 Q rows (K = 128 -> 4 chunks of 32)
  v16bf qa[4];
#pragma unroll
  for (int ks = 0; ks < 4; ++ks) qa[ks] = packA(Qp + (size_t)q0 * DKK + ks * 32, DKK, lane);

  float mstate[8], lstate[8];
  v8f oacc[8];
#pragma unroll
  for (int i = 0; i < 8; ++i) { mstate[i] = -1.0e30f; lstate[i] = 0.0f; }
#pragma unroll
  for (int nt = 0; nt < 8; ++nt) oacc[nt] = (v8f)0.0f;

  for (int kt = 0; kt < SS / 64; ++kt) {
    const int keybase = kt * 64;
    // ---- scores: S = Q K^T over 64 keys (Q pre-scaled by 1/sqrt(DK)) ----
    v8f s[4];
#pragma unroll
    for (int nt = 0; nt < 4; ++nt) s[nt] = (v8f)0.0f;
#pragma unroll
    for (int ks = 0; ks < 4; ++ks) {
#pragma unroll
      for (int nt = 0; nt < 4; ++nt) {
        v16bf bk = packB(Kp + (size_t)(keybase + nt * 16) * DKK + ks * 32, DKK, lane);
        s[nt] = wmma_bf16(qa[ks], bk, s[nt]);
      }
    }
    // ---- online softmax (row = i + 8*halfsel, striped over 16-lane groups) ----
#pragma unroll
    for (int i = 0; i < 8; ++i) {
      float tm = fmaxf(fmaxf(s[0][i], s[1][i]), fmaxf(s[2][i], s[3][i]));
#pragma unroll
      for (int off = 8; off >= 1; off >>= 1) tm = fmaxf(tm, __shfl_xor(tm, off, 32));
      float newm  = fmaxf(mstate[i], tm);
      float alpha = __expf(mstate[i] - newm);
      mstate[i] = newm;
      float p0 = __expf(s[0][i] - newm);
      float p1 = __expf(s[1][i] - newm);
      float p2 = __expf(s[2][i] - newm);
      float p3 = __expf(s[3][i] - newm);
      float ps = (p0 + p1) + (p2 + p3);
#pragma unroll
      for (int off = 8; off >= 1; off >>= 1) ps += __shfl_xor(ps, off, 32);
      lstate[i] = lstate[i] * alpha + ps;
#pragma unroll
      for (int nt = 0; nt < 8; ++nt) oacc[nt][i] *= alpha;
      int row = i + 8 * halfsel;
      Plds[w][row * PLD +      nloc] = f2bf(p0);
      Plds[w][row * PLD + 16 + nloc] = f2bf(p1);
      Plds[w][row * PLD + 32 + nloc] = f2bf(p2);
      Plds[w][row * PLD + 48 + nloc] = f2bf(p3);
    }
    asm volatile("s_wait_dscnt 0x0" ::: "memory");   // wave-private LDS round-trip
    // ---- O += P V  (K = 64 keys; B operand straight from V^T rows) ----
    v16bf pa0 = packA(&Plds[w][0],  PLD, lane);
    v16bf pa1 = packA(&Plds[w][32], PLD, lane);
#pragma unroll
    for (int nt = 0; nt < 8; ++nt) {
      v16bf bv0 = packB(vTp + (size_t)(nt * 16) * SS + keybase,      SS, lane);
      v16bf bv1 = packB(vTp + (size_t)(nt * 16) * SS + keybase + 32, SS, lane);
      oacc[nt] = wmma_bf16(pa0, bv0, oacc[nt]);
      oacc[nt] = wmma_bf16(pa1, bv1, oacc[nt]);
    }
  }
  // ---- normalize and store O as bf16 into (B*S, D) ----
#pragma unroll
  for (int i = 0; i < 8; ++i) {
    float inv = 1.0f / lstate[i];
    int row = i + 8 * halfsel;
    size_t orow = ((size_t)bb * SS + q0 + row) * DD + h * DKK;
#pragma unroll
    for (int nt = 0; nt < 8; ++nt)
      Ob[orow + nt * 16 + nloc] = f2bf(oacc[nt][i] * inv);
  }
}

// ---------------- host side ----------------
extern "C" void kernel_launch(void* const* d_in, const int* in_sizes, int n_in,
                              void* d_out, int out_size, void* d_ws, size_t ws_size,
                              hipStream_t stream) {
  const float* x     = (const float*)d_in[0];
  // d_in[1] = mask (all-ones; never applied in the reference math)
  const float* w_qkv = (const float*)d_in[2];
  const float* b_qkv = (const float*)d_in[3];
  const float* w_o   = (const float*)d_in[4];
  const float* b_o   = (const float*)d_in[5];
  float* out = (float*)d_out;

  // workspace layout (bf16/ushort units)
  unsigned short* ws = (unsigned short*)d_ws;
  const size_t N_X   = (size_t)MROWS * DD;        // 8,388,608
  const size_t N_WQ  = (size_t)E3D * DD;          // 12,582,912
  const size_t N_WO  = (size_t)DD * DD;           // 4,194,304
  const size_t N_QKV = (size_t)BB * HH * SS * DKK;// 8,388,608
  unsigned short* xb  = ws;
  unsigned short* wqb = xb  + N_X;
  unsigned short* wob = wqb + N_WQ;
  unsigned short* Qb  = wob + N_WO;
  unsigned short* Kb  = Qb  + N_QKV;
  unsigned short* vTb = Kb  + N_QKV;
  unsigned short* Ob  = vTb + N_QKV;              // total ~112 MB

  const int TB = 256;
  cvt_bf16_kernel<<<4096, TB, 0, stream>>>(x,     xb,  (int)N_X);
  cvt_bf16_kernel<<<4096, TB, 0, stream>>>(w_qkv, wqb, (int)N_WQ);
  cvt_bf16_kernel<<<4096, TB, 0, stream>>>(w_o,   wob, (int)N_WO);

  qkv_gemm_kernel<<<dim3(E3D / 128, MROWS / 256), TB, 0, stream>>>(xb, wqb, b_qkv, Qb, Kb, vTb);
  attn_kernel<<<dim3(SS / 128, BB * HH), TB, 0, stream>>>(Qb, Kb, vTb, Ob);
  oproj_gemm_kernel<<<dim3(DD / 128, MROWS / 256), TB, 0, stream>>>(Ob, wob, b_o, out);
}
